// PropeDotProductAttention_54400055771274
// MI455X (gfx1250) — matrix operand order repaired and checked by reading, not compiled, not run on hardware
//
#include <hip/hip_runtime.h>
#include <hip/hip_bf16.h>

typedef __attribute__((ext_vector_type(16))) _Float16 v16h;
typedef __attribute__((ext_vector_type(8)))  _Float16 v8h;
typedef __attribute__((ext_vector_type(8)))  float    v8f;

#define NH    16
#define SEQ   2304
#define DD    128
#define SPC   576          // S per camera
#define KCH   64           // kpos chunk per iteration
#define KROWH 136          // padded halves per K row in LDS (128 + 8)
#define VROWH 72           // padded halves per Vt row in LDS (64 + 8)
// 1/sqrt(128) * log2(e): scores produced directly in log2 domain
#define QSCALE (0.08838834764831845f * 1.4426950408889634f)

#if __has_builtin(__builtin_amdgcn_global_load_async_to_lds_b128)
#define HAS_ASYNC_LDS 1
#else
#define HAS_ASYNC_LDS 0
#endif

#if HAS_ASYNC_LDS
typedef int i32x4 __attribute__((vector_size(16)));
typedef __attribute__((address_space(1))) i32x4* as1_i32x4p;
typedef __attribute__((address_space(3))) i32x4* as3_i32x4p;
__device__ __forceinline__ void wait_asynccnt0() {
#if __has_builtin(__builtin_amdgcn_s_wait_asynccnt)
    __builtin_amdgcn_s_wait_asynccnt(0);
#else
    asm volatile("s_wait_asynccnt 0" ::: "memory");
#endif
}
#endif

// ---------------------------------------------------------------------------
// Kernel 1: build P^T, P_inv, P per camera (48 floats per cam in ws)
// ---------------------------------------------------------------------------
__global__ void prope_mats_kernel(const float* __restrict__ vm,
                                  const float* __restrict__ Kin,
                                  float* __restrict__ mats)
{
    int c = threadIdx.x;
    if (c >= 4) return;
    const float IW = 384.f;
    const float* K = Kin + c * 9;
    const float* V = vm + c * 16;
    float kn00 = K[0] / IW, kn11 = K[4] / IW;
    float kn02 = K[2] / IW - 0.5f, kn12 = K[5] / IW - 0.5f;
    float Kn[16] = { kn00, 0.f, kn02, 0.f,
                     0.f,  kn11, kn12, 0.f,
                     0.f,  0.f,  1.f,  0.f,
                     0.f,  0.f,  0.f,  1.f };
    float P[16];
    for (int i = 0; i < 4; ++i)
        for (int j = 0; j < 4; ++j) {
            float s = 0.f;
            for (int t = 0; t < 4; ++t) s += Kn[i*4+t] * V[t*4+j];
            P[i*4+j] = s;
        }
    float Vi[16] = {0.f};
    for (int i = 0; i < 3; ++i)
        for (int j = 0; j < 3; ++j) Vi[i*4+j] = V[j*4+i];
    for (int i = 0; i < 3; ++i) {
        float s = 0.f;
        for (int j = 0; j < 3; ++j) s += V[j*4+i] * V[j*4+3];
        Vi[i*4+3] = -s;
    }
    Vi[15] = 1.f;
    float Ki[16] = {0.f};
    Ki[0] = 1.f/kn00; Ki[5] = 1.f/kn11;
    Ki[2] = -kn02/kn00; Ki[6] = -kn12/kn11;
    Ki[10] = 1.f; Ki[15] = 1.f;
    float Pi[16];
    for (int i = 0; i < 4; ++i)
        for (int j = 0; j < 4; ++j) {
            float s = 0.f;
            for (int t = 0; t < 4; ++t) s += Vi[i*4+t] * Ki[t*4+j];
            Pi[i*4+j] = s;
        }
    float* o = mats + c * 48;
    for (int i = 0; i < 4; ++i)
        for (int j = 0; j < 4; ++j) {
            o[i*4+j]      = P[j*4+i];   // P^T   (for q)
            o[16 + i*4+j] = Pi[i*4+j];  // P_inv (for k, v)
            o[32 + i*4+j] = P[i*4+j];   // P     (for output)
        }
}

// ---------------------------------------------------------------------------
// Kernel 2: apply projective transform to q/k/v + convert to f16
// (q pre-scaled by 1/sqrt(D) * log2(e) for exp2-domain softmax)
// ---------------------------------------------------------------------------
__global__ __launch_bounds__(256) void prope_xform_kernel(
    const float* __restrict__ q, const float* __restrict__ k,
    const float* __restrict__ v, const float* __restrict__ mats,
    _Float16* __restrict__ qh, _Float16* __restrict__ kh,
    _Float16* __restrict__ vh)
{
    int idx = blockIdx.x * 256 + threadIdx.x;
    const int TOT = 3 * NH * SEQ * (DD / 4);
    if (idx >= TOT) return;
    int d4 = idx & 31;
    int s  = (idx >> 5) % SEQ;
    int h  = ((idx >> 5) / SEQ) % NH;
    int t  = idx / (32 * SEQ * NH);
    const float* src = (t == 0) ? q : (t == 1) ? k : v;
    _Float16*    dst = (t == 0) ? qh : (t == 1) ? kh : vh;
    size_t base = ((size_t)h * SEQ + s) * DD + d4 * 4;
    float x0 = src[base+0], x1 = src[base+1], x2 = src[base+2], x3 = src[base+3];
    if (h < 8 && d4 >= 6 && d4 <= 10) {      // channels [24,44)
        int c = s / SPC;
        const float* M = mats + c * 48 + ((t == 0) ? 0 : 16);
        float y0 = M[0]*x0  + M[1]*x1  + M[2]*x2  + M[3]*x3;
        float y1 = M[4]*x0  + M[5]*x1  + M[6]*x2  + M[7]*x3;
        float y2 = M[8]*x0  + M[9]*x1  + M[10]*x2 + M[11]*x3;
        float y3 = M[12]*x0 + M[13]*x1 + M[14]*x2 + M[15]*x3;
        x0 = y0; x1 = y1; x2 = y2; x3 = y3;
    }
    float sc = (t == 0) ? QSCALE : 1.0f;
    dst[base+0] = (_Float16)(x0 * sc);
    dst[base+1] = (_Float16)(x1 * sc);
    dst[base+2] = (_Float16)(x2 * sc);
    dst[base+3] = (_Float16)(x3 * sc);
}

// ---------------------------------------------------------------------------
// Kernel 3: flash attention, WMMA f16, exp2-domain online softmax,
// 64-kpos chunks. K path: async global->LDS DMA (double-buffered) when
// available; V path: register-staged transpose into LDS.
// Block: 256 threads = 8 waves; each wave owns 16 q rows.
// Grid: (SEQ/128, NH). Computes S^T = K*Q^T so softmax stats are per-lane.
// ---------------------------------------------------------------------------
__global__ __launch_bounds__(256) void prope_flash_attn_kernel(
    const _Float16* __restrict__ qh, const _Float16* __restrict__ kh,
    const _Float16* __restrict__ vh, float* __restrict__ out)
{
    __shared__ _Float16 Ks[2][KCH * KROWH];  // K chunk ping-pong [kpos][d]
    __shared__ _Float16 Vt[DD * VROWH];      // V chunk, transposed [d][kpos]

    const int h    = blockIdx.y;
    const int qt   = blockIdx.x;
    const int tid  = threadIdx.x;
    const int wave = tid >> 5;
    const int lane = tid & 31;
    const int ln16 = lane & 15;
    const int hig  = lane >> 4;              // lane-half (0/1)

    const int q0   = qt * 128 + wave * 16;
    const int qrow = q0 + ln16;

    // Q^T B-fragments: lane holds q-col = ln16, contiguous 16 d per half-group
    v16h qb[4];
    {
        const _Float16* qp = qh + ((size_t)h * SEQ + qrow) * DD + hig * 16;
        #pragma unroll
        for (int dc = 0; dc < 4; ++dc)
            qb[dc] = *(const v16h*)(qp + dc * 32);
    }

    const int krow_t = tid >> 4;             // 0..15 base row index helper
    const int c8     = tid & 15;             // 8-half column group
    v8h vreg[4];
#if !HAS_ASYNC_LDS
    v8h kreg[4];
#endif

    auto fetch_k = [&](int kc, int buf) {
        const int k0 = kc * KCH;
        #pragma unroll
        for (int i = 0; i < 4; ++i) {
            int krow = krow_t + i * 16;      // 0..63
            const _Float16* g =
                kh + ((size_t)h * SEQ + k0 + krow) * DD + c8 * 8;
#if HAS_ASYNC_LDS
            __builtin_amdgcn_global_load_async_to_lds_b128(
                (as1_i32x4p)g,
                (as3_i32x4p)&Ks[buf][krow * KROWH + c8 * 8],
                0, 0);
#else
            kreg[i] = *(const v8h*)g;
#endif
        }
    };
    auto fetch_v = [&](int kc) {
        const int k0 = kc * KCH;
        #pragma unroll
        for (int i = 0; i < 4; ++i) {
            int krow = krow_t + i * 16;
            vreg[i] = *(const v8h*)(vh + ((size_t)h * SEQ + k0 + krow) * DD
                                    + c8 * 8);
        }
    };
    auto deposit = [&](int buf) {            // regs -> LDS
        #pragma unroll
        for (int i = 0; i < 4; ++i) {
            int krow = krow_t + i * 16;
#if !HAS_ASYNC_LDS
            *(v8h*)(&Ks[buf][krow * KROWH + c8 * 8]) = kreg[i];
#endif
            #pragma unroll
            for (int j = 0; j < 8; ++j)
                Vt[(c8 * 8 + j) * VROWH + krow] = vreg[i][j];
        }
    };

    v8f acc[8];
    #pragma unroll
    for (int i = 0; i < 8; ++i) { v8f z = {}; acc[i] = z; }
    float m = -1e30f, l = 0.f;

    const int NIT = SEQ / KCH;
    fetch_k(0, 0);
    fetch_v(0);

    for (int kc = 0; kc < NIT; ++kc) {
        const int cur = kc & 1;
        __syncthreads();                     // prev chunk's consumers done
        deposit(cur);
#if HAS_ASYNC_LDS
        wait_asynccnt0();                    // K DMA into Ks[cur] complete
#endif
        __syncthreads();                     // Ks[cur] + Vt visible to all
        if (kc + 1 < NIT) {
            fetch_k(kc + 1, cur ^ 1);        // DMA next K while computing
            fetch_v(kc + 1);                 // next V into regs
        }
        const _Float16* ks = &Ks[cur][0];

        // S^T tiles: [kpos x q], 4 kpos tiles of 16
        v8f st[4];
        #pragma unroll
        for (int t4 = 0; t4 < 4; ++t4) {
            v8f s = {};
            const int krow = t4 * 16 + ln16;
            #pragma unroll
            for (int dc = 0; dc < 4; ++dc) {
                const _Float16* kp = ks + krow * KROWH + dc * 32 + hig * 16;
                v8h alo = *(const v8h*)(kp);
                v8h ahi = *(const v8h*)(kp + 8);
                v16h a = {};
                #pragma unroll
                for (int j = 0; j < 8; ++j) { a[j] = alo[j]; a[8 + j] = ahi[j]; }
                s = __builtin_amdgcn_wmma_f32_16x16x32_f16(
                        false, a, false, qb[dc], (short)0, s, false, false);
            }
            st[t4] = s;
        }

        // online softmax in log2 domain (stats per q = ln16)
        float cmx = -1e30f;
        #pragma unroll
        for (int t4 = 0; t4 < 4; ++t4)
            #pragma unroll
            for (int i = 0; i < 8; ++i)
                cmx = fmaxf(cmx, st[t4][i]);
        cmx = fmaxf(cmx, __shfl_xor(cmx, 16, 32));
        float mnew  = fmaxf(m, cmx);
        float alpha = __builtin_amdgcn_exp2f(m - mnew);
        float rsum  = 0.f;
        #pragma unroll
        for (int t4 = 0; t4 < 4; ++t4)
            #pragma unroll
            for (int i = 0; i < 8; ++i) {
                st[t4][i] = __builtin_amdgcn_exp2f(st[t4][i] - mnew);
                rsum += st[t4][i];
            }
        rsum += __shfl_xor(rsum, 16, 32);
        l = l * alpha + rsum;
        m = mnew;

        // rescale accumulators once per 64 kpos: row r+8*hig uses lane r+8*hig
        #pragma unroll
        for (int r = 0; r < 8; ++r) {
            float ar = __shfl(alpha, r + 8 * hig, 32);
            #pragma unroll
            for (int dt = 0; dt < 8; ++dt) acc[dt][r] *= ar;
        }

        // out += P * V, two 32-kpos sub-chunks
        #pragma unroll
        for (int cc = 0; cc < 2; ++cc) {
            v16h p = {};
            #pragma unroll
            for (int i = 0; i < 8; ++i) {
                p[i]     = (_Float16)st[2*cc][i];
                p[8 + i] = (_Float16)st[2*cc + 1][i];
            }
            #pragma unroll
            for (int dt = 0; dt < 8; ++dt) {
                const _Float16* vp =
                    &Vt[(dt * 16 + ln16) * VROWH + cc * 32 + hig * 8];
                v8h blo = *(const v8h*)(vp);
                v8h bhi = *(const v8h*)(vp + 16);
                v16h b = {};
                #pragma unroll
                for (int j = 0; j < 8; ++j) { b[j] = blo[j]; b[8 + j] = bhi[j]; }
                acc[dt] = __builtin_amdgcn_wmma_f32_16x16x32_f16(
                              false, p, false, b, (short)0, acc[dt], false, false);
            }
        }
    }

    // finalize: divide by l, write f32
    float linv = 1.f / l;
    #pragma unroll
    for (int r = 0; r < 8; ++r) {
        float lr  = __shfl(linv, r + 8 * hig, 32);
        int   row = q0 + r + 8 * hig;
        float* op = out + ((size_t)h * SEQ + row) * DD + ln16;
        #pragma unroll
        for (int dt = 0; dt < 8; ++dt)
            op[dt * 16] = acc[dt][r] * lr;
    }
}

// ---------------------------------------------------------------------------
// Kernel 4: apply P to channels [24,44) of heads 0..7 of output, in place
// ---------------------------------------------------------------------------
__global__ __launch_bounds__(256) void prope_outproj_kernel(
    float* __restrict__ out, const float* __restrict__ mats)
{
    int idx = blockIdx.x * 256 + threadIdx.x;
    const int TOT = 8 * SEQ * 5;
    if (idx >= TOT) return;
    int g = idx % 5;
    int s = (idx / 5) % SEQ;
    int h = idx / (5 * SEQ);
    int c = s / SPC;
    const float* M = mats + c * 48 + 32;   // P
    float* p = out + ((size_t)h * SEQ + s) * DD + 24 + g * 4;
    float x0 = p[0], x1 = p[1], x2 = p[2], x3 = p[3];
    p[0] = M[0]*x0  + M[1]*x1  + M[2]*x2  + M[3]*x3;
    p[1] = M[4]*x0  + M[5]*x1  + M[6]*x2  + M[7]*x3;
    p[2] = M[8]*x0  + M[9]*x1  + M[10]*x2 + M[11]*x3;
    p[3] = M[12]*x0 + M[13]*x1 + M[14]*x2 + M[15]*x3;
}

// ---------------------------------------------------------------------------
extern "C" void kernel_launch(void* const* d_in, const int* in_sizes, int n_in,
                              void* d_out, int out_size, void* d_ws, size_t ws_size,
                              hipStream_t stream)
{
    const float* q  = (const float*)d_in[0];
    const float* k  = (const float*)d_in[1];
    const float* v  = (const float*)d_in[2];
    const float* vm = (const float*)d_in[3];
    const float* Ks = (const float*)d_in[4];
    float* out = (float*)d_out;

    char* ws = (char*)d_ws;
    float*    mats = (float*)ws;                         // 768 B (pad 1 KiB)
    _Float16* qh   = (_Float16*)(ws + 1024);
    _Float16* kh   = qh + (size_t)NH * SEQ * DD;
    _Float16* vh   = kh + (size_t)NH * SEQ * DD;

    prope_mats_kernel<<<1, 32, 0, stream>>>(vm, Ks, mats);

    int tot = 3 * NH * SEQ * (DD / 4);
    prope_xform_kernel<<<(tot + 255) / 256, 256, 0, stream>>>(q, k, v, mats,
                                                              qh, kh, vh);

    dim3 grid(SEQ / 128, NH);
    prope_flash_attn_kernel<<<grid, 256, 0, stream>>>(qh, kh, vh, out);

    int tot2 = 8 * SEQ * 5;
    prope_outproj_kernel<<<(tot2 + 255) / 256, 256, 0, stream>>>(out, mats);
}
